// PositionalSpatialAttention_39470749450436
// MI455X (gfx1250) — matrix operand order repaired and checked by reading, not compiled, not run on hardware
//
#include <hip/hip_runtime.h>
#include <hip/hip_bf16.h>

typedef __attribute__((ext_vector_type(16))) __bf16 v16bf;
typedef __attribute__((ext_vector_type(8)))  float  v8f;

#define B_ 16
#define C_ 256
#define D_ 64
#define N_ 4096
#define SCALE_ 0.125f   // 1/sqrt(64), TEMP=1

union BF16x16 { v16bf v; unsigned u[8]; uint4 q[2]; };

__device__ __forceinline__ unsigned pk(float a, float b) {
  unsigned ua = __float_as_uint(a), ub = __float_as_uint(b);
  ua = (ua + 0x7FFFu + ((ua >> 16) & 1u)) >> 16;
  ub = (ub + 0x7FFFu + ((ub >> 16) & 1u)) >> 16;
  return (ua & 0xFFFFu) | (ub << 16);
}

__device__ __forceinline__ v8f wmma_bf16(const BF16x16& a, const BF16x16& b, v8f c) {
  return __builtin_amdgcn_wmma_f32_16x16x32_bf16(false, a.v, false, b.v,
                                                 (short)0, c, false, false);
}

// ---- CDNA5 async global->LDS copy (ASYNCcnt-tracked DMA path) ----
typedef __attribute__((address_space(3))) unsigned char lds_t;
__device__ __forceinline__ unsigned ldsoff(void* p) {
  return (unsigned)(unsigned long long)(lds_t*)p;
}
__device__ __forceinline__ void async_b128(unsigned lds_addr, const void* gaddr) {
  asm volatile("global_load_async_to_lds_b128 %0, %1, off"
               :: "v"(lds_addr), "v"(gaddr) : "memory");
}
__device__ __forceinline__ void wait_async0() {
  asm volatile("s_wait_asynccnt 0x0" ::: "memory");
}

// ---- workspace layout (bytes) ----
constexpr size_t XP_OFF  = 0;
constexpr size_t XP_SZ   = (size_t)B_ * 256 * 8 * 32 * 32;     // x  A-operand packed
constexpr size_t QP_OFF  = XP_OFF + XP_SZ;
constexpr size_t QP_SZ   = (size_t)B_ * 256 * 2 * 32 * 32;     // q  A-operand (pre-scaled)
constexpr size_t KP_OFF  = QP_OFF + QP_SZ;
constexpr size_t KP_SZ   = QP_SZ;                              // k  B-operand (K-dim=d)
constexpr size_t VP_OFF  = KP_OFF + KP_SZ;
constexpr size_t VP_SZ   = (size_t)B_ * 128 * 4 * 32 * 32;     // v  B-operand (K-dim=keys)
constexpr size_t WQKV_OFF= VP_OFF + VP_SZ;
constexpr size_t WQKV_SZ = (size_t)3 * 32 * 32 * 32;
constexpr size_t WO_OFF  = WQKV_OFF + WQKV_SZ;
constexpr size_t WO_SZ   = (size_t)32 * 32 * 32;
constexpr size_t PT_OFF  = WO_OFF + WO_SZ;                     // posT[n][d] fp32

// grid segmentation for pack_kernel
constexpr int XPB  = B_ * 256 * 8;   // 32768
constexpr int WQB  = XPB + 96;
constexpr int WOB  = WQB + 32;
constexpr int POSB = WOB + 256;

// =================== kernel 1: convert + pre-pack operands ===================
__global__ __launch_bounds__(32) void pack_kernel(const float* __restrict__ x,
                            const float* __restrict__ pos,
                            const float* __restrict__ wq,
                            const float* __restrict__ wk,
                            const float* __restrict__ wv,
                            const float* __restrict__ wo,
                            char* __restrict__ ws) {
  const int bx   = blockIdx.x;
  const int lane = threadIdx.x;
  const int col  = lane & 15;
  const int half = lane >> 4;
  unsigned dw[8];

  if (bx < XPB) {
    // xPack: A layout 16x32 bf16 (rows=n, K=c)
    const int b  = bx >> 11;
    const int nt = (bx >> 3) & 255;
    const int cc = bx & 7;
    const int n  = nt * 16 + col;
    const float* xb = x + (size_t)b * C_ * N_;
    const int cbase = cc * 32 + half * 16;
#pragma unroll
    for (int t = 0; t < 4; ++t) {
      dw[t]     = pk(xb[(size_t)(cbase + 2*t) * N_ + n],
                     xb[(size_t)(cbase + 2*t + 1) * N_ + n]);
      dw[4 + t] = pk(xb[(size_t)(cbase + 8 + 2*t) * N_ + n],
                     xb[(size_t)(cbase + 8 + 2*t + 1) * N_ + n]);
    }
    uint4* dst = (uint4*)(ws + XP_OFF + ((size_t)bx * 32 + lane) * 32);
    dst[0] = make_uint4(dw[0], dw[1], dw[2], dw[3]);
    dst[1] = make_uint4(dw[4], dw[5], dw[6], dw[7]);
  } else if (bx < WQB) {
    // wq/wk/wv B layout 32x16 (K=c, col=d)
    const int g  = bx - XPB;                 // m*32 + dt*8 + cc
    const int m  = g >> 5;
    const int dt = (g >> 3) & 3;
    const int cc = g & 7;
    const float* w = (m == 0) ? wq : (m == 1) ? wk : wv;
    const int d = dt * 16 + col;
    const float* row = w + (size_t)d * C_ + cc * 32 + half * 16;
#pragma unroll
    for (int t = 0; t < 8; ++t) dw[t] = pk(row[2*t], row[2*t + 1]);
    uint4* dst = (uint4*)(ws + WQKV_OFF + ((size_t)g * 32 + lane) * 32);
    dst[0] = make_uint4(dw[0], dw[1], dw[2], dw[3]);
    dst[1] = make_uint4(dw[4], dw[5], dw[6], dw[7]);
  } else if (bx < WOB) {
    // wo^T B layout (K=d, col=c)
    const int g  = bx - WQB;                 // ct*2 + dc
    const int ct = g >> 1;
    const int dc = g & 1;
    const int c  = ct * 16 + col;
    const float* row = wo + (size_t)c * D_ + dc * 32 + half * 16;
#pragma unroll
    for (int t = 0; t < 8; ++t) dw[t] = pk(row[2*t], row[2*t + 1]);
    uint4* dst = (uint4*)(ws + WO_OFF + ((size_t)g * 32 + lane) * 32);
    dst[0] = make_uint4(dw[0], dw[1], dw[2], dw[3]);
    dst[1] = make_uint4(dw[4], dw[5], dw[6], dw[7]);
  } else {
    // posT[n][d] = pos[d][n]  (reads strided by N across lanes, coalesced writes along d)
    const int nb = bx - WOB;                 // 0..255, 16 rows each
    const int d0 = lane * 2;
    float* posT = (float*)(ws + PT_OFF);
#pragma unroll
    for (int i = 0; i < 16; ++i) {
      const int n = nb * 16 + i;
      float a  = pos[(size_t)d0 * N_ + n];
      float c2 = pos[(size_t)(d0 + 1) * N_ + n];
      *(float2*)(posT + (size_t)n * 64 + d0) = make_float2(a, c2);
    }
  }
}

// =================== kernel 2: QKV projection + repack ===================
__global__ __launch_bounds__(64) void qkv_kernel(const float* __restrict__ bq,
                           const float* __restrict__ bk,
                           const float* __restrict__ bvp,
                           char* __restrict__ ws) {
  __shared__ float stage[32 * 64];
  const int b    = blockIdx.x >> 7;
  const int blk  = blockIdx.x & 127;          // 32-row block
  const int w    = threadIdx.x >> 5;
  const int lane = threadIdx.x & 31;
  const int col  = lane & 15;
  const int half = lane >> 4;
  const int nt   = blk * 2 + w;               // this wave's 16-row tile
  const int n0   = nt * 16;
  const uint4* xp = (const uint4*)(ws + XP_OFF);
  const uint4* wp = (const uint4*)(ws + WQKV_OFF);
  const float* posT = (const float*)(ws + PT_OFF);
  unsigned dw[8];

  for (int m = 0; m < 3; ++m) {
    v8f acc[4] = {};
    for (int cc = 0; cc < 8; ++cc) {
      BF16x16 A;
      size_t ab = (((size_t)(b * 256 + nt) * 8 + cc) * 32 + lane) * 2;
      A.q[0] = xp[ab]; A.q[1] = xp[ab + 1];
#pragma unroll
      for (int dt = 0; dt < 4; ++dt) {
        BF16x16 Bm;
        size_t bb = (((size_t)(m * 32 + dt * 8 + cc)) * 32 + lane) * 2;
        Bm.q[0] = wp[bb]; Bm.q[1] = wp[bb + 1];
        acc[dt] = wmma_bf16(A, Bm, acc[dt]);
      }
    }
    const float* bias = (m == 0) ? bq : (m == 1) ? bk : bvp;
#pragma unroll
    for (int dt = 0; dt < 4; ++dt) {
      const int d = dt * 16 + col;
      const float bb = bias[d];
#pragma unroll
      for (int j = 0; j < 8; ++j) {
        const int n = n0 + j + 8 * half;
        float v = acc[dt][j] + bb;
        if (m < 2) v += posT[(size_t)n * 64 + d];
        if (m == 0) v *= SCALE_;              // fold 1/sqrt(d) into q
        stage[(w * 16 + j + 8 * half) * 64 + d] = v;
      }
    }
    __syncthreads();
    if (m == 0) {            // q -> A layout (rows=q, K=d)
#pragma unroll
      for (int c = 0; c < 2; ++c) {
        const float* r = &stage[(w * 16 + col) * 64 + c * 32 + half * 16];
#pragma unroll
        for (int t = 0; t < 4; ++t) {
          dw[t]     = pk(r[2*t],      r[2*t + 1]);
          dw[4 + t] = pk(r[8 + 2*t], r[8 + 2*t + 1]);
        }
        uint4* dst = (uint4*)(ws + QP_OFF +
                              (((size_t)(b * 256 + nt) * 2 + c) * 32 + lane) * 32);
        dst[0] = make_uint4(dw[0], dw[1], dw[2], dw[3]);
        dst[1] = make_uint4(dw[4], dw[5], dw[6], dw[7]);
      }
    } else if (m == 1) {     // k -> B layout for QK^T (K=d, col=key)
#pragma unroll
      for (int c = 0; c < 2; ++c) {
        const float* r = &stage[(w * 16 + col) * 64 + c * 32 + half * 16];
#pragma unroll
        for (int t = 0; t < 8; ++t) dw[t] = pk(r[2*t], r[2*t + 1]);
        uint4* dst = (uint4*)(ws + KP_OFF +
                              (((size_t)(b * 256 + nt) * 2 + c) * 32 + lane) * 32);
        dst[0] = make_uint4(dw[0], dw[1], dw[2], dw[3]);
        dst[1] = make_uint4(dw[4], dw[5], dw[6], dw[7]);
      }
    } else {                 // v -> B layout for PV (K=key(32), col=d)
#pragma unroll
      for (int dth = 0; dth < 2; ++dth) {
        const int dt = w * 2 + dth;
        const int d  = dt * 16 + col;
        const int kb = half * 16;
#pragma unroll
        for (int t = 0; t < 8; ++t)
          dw[t] = pk(stage[(kb + 2*t) * 64 + d], stage[(kb + 2*t + 1) * 64 + d]);
        uint4* dst = (uint4*)(ws + VP_OFF +
                              (((size_t)(b * 128 + blk) * 4 + dt) * 32 + lane) * 32);
        dst[0] = make_uint4(dw[0], dw[1], dw[2], dw[3]);
        dst[1] = make_uint4(dw[4], dw[5], dw[6], dw[7]);
      }
    }
    __syncthreads();
  }
}

// ========== kernel 3: flash attention (async LDS K/V) + out proj + residual ==========
__global__ __launch_bounds__(256) void attn_kernel(const float* __restrict__ x,
                            const float* __restrict__ bo,
                            const float* __restrict__ gamma,
                            float* __restrict__ out,
                            char* __restrict__ ws) {
  __shared__ float smem[8192];                 // 32 KB
  const int b    = blockIdx.x >> 5;            // grid = B*32
  const int grp  = blockIdx.x & 31;
  const int tid  = threadIdx.x;
  const int w    = tid >> 5;
  const int lane = tid & 31;
  const int col  = lane & 15;
  const int half = lane >> 4;
  const int nt   = grp * 8 + w;
  const int n0   = nt * 16;
  uint4* kvbuf0 = (uint4*)smem;                // 8KB tile: 4 K groups + 4 V groups
  uint4* kvbuf1 = (uint4*)(smem + 2048);
  float* pbuf   = smem + 4096 + w * 512;       // per-wave P scratch (16x32)
  const uint4* qp  = (const uint4*)(ws + QP_OFF);
  const uint4* kp  = (const uint4*)(ws + KP_OFF);
  const uint4* vp  = (const uint4*)(ws + VP_OFF);
  const uint4* wop = (const uint4*)(ws + WO_OFF);

  // cooperative KV copy mapping: tid<128 -> K groups (kt,c), else V groups (dt)
  const int  cg   = (tid >> 5) & 3;
  const bool isK  = tid < 128;
  const int  dsti = ((isK ? cg : 4 + cg) * 32 + lane) * 2;
  const uint4* srcp = isK ? kp : vp;
  const unsigned dst0 = ldsoff(&kvbuf0[dsti]); // per-thread LDS byte addrs
  const unsigned dst1 = ldsoff(&kvbuf1[dsti]);

  BF16x16 QA[2];
#pragma unroll
  for (int c = 0; c < 2; ++c) {
    size_t qb = (((size_t)(b * 256 + nt) * 2 + c) * 32 + lane) * 2;
    QA[c].q[0] = qp[qb]; QA[c].q[1] = qp[qb + 1];
  }
  BF16x16 ONES;
#pragma unroll
  for (int t = 0; t < 8; ++t) ONES.u[t] = 0x3F803F80u;

  // prologue: async-DMA tile 0 into kvbuf0
  {
    size_t s = isK ? (((size_t)(b * 256 + (cg >> 1)) * 2 + (cg & 1)) * 32 + lane) * 2
                   : (((size_t)(b * 128) * 4 + cg) * 32 + lane) * 2;
    async_b128(dst0,      srcp + s);
    async_b128(dst0 + 16, srcp + s + 1);
  }
  wait_async0();
  __syncthreads();

  float mrow[8];
#pragma unroll
  for (int j = 0; j < 8; ++j) mrow[j] = -3.0e38f;
  v8f O[5] = {};                               // O[4] = softmax denominator (ones column)

  for (int kb = 0; kb < 128; ++kb) {
    const uint4* cur = (kb & 1) ? kvbuf1 : kvbuf0;
    const unsigned nxt = (kb & 1) ? dst0 : dst1;
    if (kb + 1 < 128) {                        // background DMA of next tile
      size_t s = isK ? (((size_t)(b * 256 + (kb + 1) * 2 + (cg >> 1)) * 2 + (cg & 1)) * 32 + lane) * 2
                     : (((size_t)(b * 128 + kb + 1) * 4 + cg) * 32 + lane) * 2;
      async_b128(nxt,      srcp + s);
      async_b128(nxt + 16, srcp + s + 1);
    }
    // S = qK^T   (q pre-scaled)
    v8f S0 = {}, S1 = {};
#pragma unroll
    for (int c = 0; c < 2; ++c) {
      BF16x16 K0, K1;
      const int i0 = ((0 * 2 + c) * 32 + lane) * 2;
      const int i1 = ((1 * 2 + c) * 32 + lane) * 2;
      K0.q[0] = cur[i0]; K0.q[1] = cur[i0 + 1];
      K1.q[0] = cur[i1]; K1.q[1] = cur[i1 + 1];
      S0 = wmma_bf16(QA[c], K0, S0);
      S1 = wmma_bf16(QA[c], K1, S1);
    }
    float alpha[8];
#pragma unroll
    for (int j = 0; j < 8; ++j) {
      float s0 = S0[j], s1 = S1[j];
      float mx = fmaxf(s0, s1);
      mx = fmaxf(mx, __shfl_xor(mx, 1, 32));
      mx = fmaxf(mx, __shfl_xor(mx, 2, 32));
      mx = fmaxf(mx, __shfl_xor(mx, 4, 32));
      mx = fmaxf(mx, __shfl_xor(mx, 8, 32));
      float mnew = fmaxf(mrow[j], mx);
      alpha[j] = __expf(mrow[j] - mnew);
      mrow[j] = mnew;
      float p0 = __expf(s0 - mnew), p1 = __expf(s1 - mnew);
      const int row = j + 8 * half;
      pbuf[row * 32 + col]      = p0;
      pbuf[row * 32 + 16 + col] = p1;
    }
#pragma unroll
    for (int dt = 0; dt < 5; ++dt)
#pragma unroll
      for (int j = 0; j < 8; ++j) O[dt][j] *= alpha[j];
    BF16x16 PA;
    {
      const float* r = &pbuf[col * 32 + half * 16];
#pragma unroll
      for (int t = 0; t < 4; ++t) {
        PA.u[t]     = pk(r[2*t],      r[2*t + 1]);
        PA.u[4 + t] = pk(r[8 + 2*t], r[8 + 2*t + 1]);
      }
    }
#pragma unroll
    for (int dt = 0; dt < 4; ++dt) {
      BF16x16 VB;
      const int vi = ((4 + dt) * 32 + lane) * 2;
      VB.q[0] = cur[vi]; VB.q[1] = cur[vi + 1];
      O[dt] = wmma_bf16(PA, VB, O[dt]);
    }
    O[4] = wmma_bf16(PA, ONES, O[4]);          // row-sum accumulator
    wait_async0();                             // next tile resident in LDS
    __syncthreads();
  }

  // finalize: attended = O / l  (l = ones-column accumulator), stage per-wave
  float* obuf = smem + w * 1024;               // overlay (all waves past final barrier)
  float inv[8];
#pragma unroll
  for (int j = 0; j < 8; ++j) inv[j] = 1.0f / O[4][j];
#pragma unroll
  for (int dt = 0; dt < 4; ++dt)
#pragma unroll
    for (int j = 0; j < 8; ++j)
      obuf[(j + 8 * half) * 64 + dt * 16 + col] = O[dt][j] * inv[j];
  BF16x16 OA[2];
#pragma unroll
  for (int c = 0; c < 2; ++c) {
    const float* r = &obuf[col * 64 + c * 32 + half * 16];
#pragma unroll
    for (int t = 0; t < 4; ++t) {
      OA[c].u[t]     = pk(r[2*t],      r[2*t + 1]);
      OA[c].u[4 + t] = pk(r[8 + 2*t], r[8 + 2*t + 1]);
    }
  }
  const float g = gamma[0];
  for (int ct = 0; ct < 16; ++ct) {
    v8f acc = {};
#pragma unroll
    for (int dc = 0; dc < 2; ++dc) {
      BF16x16 WB;
      size_t wb = (((size_t)(ct * 2 + dc)) * 32 + lane) * 2;
      WB.q[0] = wop[wb]; WB.q[1] = wop[wb + 1];
      acc = wmma_bf16(OA[dc], WB, acc);
    }
    const int c = ct * 16 + col;
    const float bc = bo[c];
#pragma unroll
    for (int j = 0; j < 8; ++j)
      obuf[(j + 8 * half) * 16 + col] = g * (acc[j] + bc);   // stage tile
    // coalesced residual + store: lane -> (c_local = lane/2, n run of 8)
    const int cl = lane >> 1;
    const int nr = (lane & 1) * 8;
    size_t base = ((size_t)b * C_ + ct * 16 + cl) * N_ + n0 + nr;
    float4 x0 = *(const float4*)(x + base);
    float4 x1 = *(const float4*)(x + base + 4);
    float4 o0, o1;
    o0.x = obuf[(nr + 0) * 16 + cl] + x0.x;
    o0.y = obuf[(nr + 1) * 16 + cl] + x0.y;
    o0.z = obuf[(nr + 2) * 16 + cl] + x0.z;
    o0.w = obuf[(nr + 3) * 16 + cl] + x0.w;
    o1.x = obuf[(nr + 4) * 16 + cl] + x1.x;
    o1.y = obuf[(nr + 5) * 16 + cl] + x1.y;
    o1.z = obuf[(nr + 6) * 16 + cl] + x1.z;
    o1.w = obuf[(nr + 7) * 16 + cl] + x1.w;
    *(float4*)(out + base)     = o0;
    *(float4*)(out + base + 4) = o1;
  }
}

extern "C" void kernel_launch(void* const* d_in, const int* in_sizes, int n_in,
                              void* d_out, int out_size, void* d_ws, size_t ws_size,
                              hipStream_t stream) {
  (void)in_sizes; (void)n_in; (void)out_size; (void)ws_size;
  const float* x     = (const float*)d_in[0];
  const float* pos   = (const float*)d_in[1];
  const float* wq    = (const float*)d_in[2];
  const float* bq    = (const float*)d_in[3];
  const float* wk    = (const float*)d_in[4];
  const float* bk    = (const float*)d_in[5];
  const float* wv    = (const float*)d_in[6];
  const float* bv    = (const float*)d_in[7];
  const float* wo    = (const float*)d_in[8];
  const float* bo    = (const float*)d_in[9];
  const float* gamma = (const float*)d_in[10];
  float* out = (float*)d_out;
  char*  ws  = (char*)d_ws;

  pack_kernel<<<POSB, 32, 0, stream>>>(x, pos, wq, wk, wv, wo, ws);
  qkv_kernel<<<B_ * 128, 64, 0, stream>>>(bq, bk, bv, ws);
  attn_kernel<<<B_ * 32, 256, 0, stream>>>(x, bo, gamma, out, ws);
}
